// TabularModel_8950711844965
// MI455X (gfx1250) — matrix operand order repaired and checked by reading, compiled, not run on hardware
//
#include <hip/hip_runtime.h>
#include <hip/hip_bf16.h>
#include <stdint.h>

typedef __bf16        v16bf __attribute__((ext_vector_type(16)));
typedef float         v8f   __attribute__((ext_vector_type(8)));
typedef unsigned int  u32x4 __attribute__((ext_vector_type(4)));

#define N_CAT   1479
#define K1      1490            // 1479 one-hot + 11 continuous
#define K1P     1504            // padded to 47*32
#define KT1     47
#define N1      200
#define NT1     13              // padded to 208
#define K2      200
#define K2P     224             // padded to 7*32
#define KT2     7
#define N2      100
#define NT2     7               // padded to 112
#define B_ROWS  65536
#define TM      64

#define WT_ELEMS  (N_CAT * K1P)          // 2,224,416 f32 (~8.9 MB)
#define W1P_ELEMS (NT1 * KT1 * 256)      // 156,416 u32  (~0.6 MB)
#define W2P_ELEMS (NT2 * KT2 * 256)      // 12,544 u32

__device__ __constant__ int c_off[12] = {0,4,16,47,54,74,424,774,1124,1474,1476,0};

static __device__ inline unsigned short f2bf(float f) {
    unsigned u = __builtin_bit_cast(unsigned, f);
    u += 0x7FFFu + ((u >> 16) & 1u);            // round-to-nearest-even
    return (unsigned short)(u >> 16);
}
static __device__ inline unsigned pack2(float a, float b) {
    return (unsigned)f2bf(a) | ((unsigned)f2bf(b) << 16);
}

// ---- prep 1: Wt[j][k] = W_cat[k][j], k-padded with zeros (coalesced gather rows) ----
__global__ void prep_wt(const float* __restrict__ W_cat, float* __restrict__ Wt) {
    int t = blockIdx.x * blockDim.x + threadIdx.x;
    if (t >= WT_ELEMS) return;
    int j = t / K1P, k = t - j * K1P;
    Wt[t] = (k < N_CAT) ? W_cat[k * N_CAT + j] : 0.f;
}

// ---- prep 2: pack W ([n][k] row-major f32) into bf16 WMMA B fragments ----
// layout: Wp[((nt*ktiles + kt)*32 + lane)*8 + j]; B 32x16: lanes0-15 K=0-7,16-23, lanes16-31 K=8-15,24-31
__global__ void prep_pack(const float* __restrict__ W, unsigned* __restrict__ Wp,
                          int ntiles, int ktiles, int nmax, int kmax, int ld) {
    int t = blockIdx.x * blockDim.x + threadIdx.x;
    if (t >= ntiles * ktiles * 256) return;
    int j    = t & 7;
    int lane = (t >> 3) & 31;
    int fidx = t >> 8;
    int kt = fidx % ktiles, nt = fidx / ktiles;
    int n = nt * 16 + (lane & 15);
    int k = kt * 32 + ((j < 4) ? ((lane >> 4) * 8 + 2 * j) : (16 + (lane >> 4) * 8 + 2 * (j - 4)));
    float a = (n < nmax && k     < kmax) ? W[n * ld + k]     : 0.f;
    float b = (n < nmax && k + 1 < kmax) ? W[n * ld + k + 1] : 0.f;
    Wp[t] = pack2(a, b);
}

// ---- fused: gather-embed + relu + MLP(1490->200->100->1), one block per 64 rows ----
__global__ __launch_bounds__(128) void fused_mlp(
    const int* __restrict__ x_cat, const float* __restrict__ x_cont,
    const float* __restrict__ b_cat, const float* __restrict__ b1,
    const float* __restrict__ b2, const float* __restrict__ W3,
    const float* __restrict__ b3, const float* __restrict__ Wt,
    const unsigned* __restrict__ W1p, const unsigned* __restrict__ W2p,
    float* __restrict__ out)
{
    __shared__ int   gidx[TM][12];                      // 12-padded -> 48B rows, b128-loadable
    __shared__ float hbuf[4][16][32];                   // per-wave h chunk (f32)
    __shared__ __align__(16) unsigned short x1b[4][16][K2P]; // layer-1 acts, bf16, k-padded

    const int tid  = threadIdx.x;
    const int lane = tid & 31;
    const int wid  = tid >> 5;
    const int half = lane >> 4;
    const int mlo  = lane & 15;
    const int row0 = blockIdx.x * TM;

    // prologue: row offsets into Wt, zero the padded activation buffer
    for (int t = tid; t < TM * 12; t += 128) {
        int r = t / 12, i = t - r * 12;
        gidx[r][i] = (i < 11) ? (x_cat[(row0 + r) * 11 + i] + c_off[i]) * K1P : 0;
    }
    for (int t = tid; t < 4 * 16 * (K2P / 2); t += 128)
        ((unsigned*)x1b)[t] = 0u;
    __syncthreads();

    union Frag { unsigned u[8]; u32x4 q[2]; v16bf v; };

    v8f acc1[NT1];
    #pragma unroll
    for (int nt = 0; nt < NT1; ++nt)
        #pragma unroll
        for (int v = 0; v < 8; ++v) acc1[nt][v] = 0.f;

    for (int kt = 0; kt < KT1; ++kt) {
        // ---- produce: lane == k; this wave's 16 rows ----
        const int kk = kt * 32 + lane;
        const float bias = (kk < N_CAT) ? b_cat[kk] : 0.f;
        for (int r = 0; r < 16; ++r) {
            const int4* gp = (const int4*)gidx[wid * 16 + r];
            int4 g0 = gp[0], g1 = gp[1], g2 = gp[2];
            float h;
            if (kk < N_CAT) {                          // one-hot gather-sum + relu
                float s = bias;
                s += Wt[g0.x + kk]; s += Wt[g0.y + kk]; s += Wt[g0.z + kk]; s += Wt[g0.w + kk];
                s += Wt[g1.x + kk]; s += Wt[g1.y + kk]; s += Wt[g1.z + kk]; s += Wt[g1.w + kk];
                s += Wt[g2.x + kk]; s += Wt[g2.y + kk]; s += Wt[g2.z + kk];
                h = fmaxf(s, 0.f);
            } else if (kk < K1) {                      // continuous features (no relu)
                h = x_cont[(row0 + wid * 16 + r) * 11 + (kk - N_CAT)];
            } else {
                h = 0.f;                               // k padding
            }
            hbuf[wid][r][lane] = h;
        }
        // same-wave LDS produce->consume: DS ops are in-order per wave, no barrier

        // ---- A fragment (16x32 bf16, ISA 7.12.2 layout) ----
        Frag a;
        #pragma unroll
        for (int j = 0; j < 8; ++j) {
            int ko = (j < 4) ? (half * 8 + 2 * j) : (16 + half * 8 + 2 * (j - 4));
            float2 p = *(const float2*)&hbuf[wid][mlo][ko];
            a.u[j] = pack2(p.x, p.y);
        }

        // ---- 13 B fragments + WMMA, 1-deep software pipeline ----
        const unsigned* wbase = W1p + ((unsigned)kt * 32u + lane) * 8u;
        Frag bcur, bnxt;
        bcur.q[0] = ((const u32x4*)wbase)[0];
        bcur.q[1] = ((const u32x4*)wbase)[1];
        #pragma unroll
        for (int nt = 0; nt < NT1; ++nt) {
            if (nt + 1 < NT1) {
                const u32x4* p = (const u32x4*)(wbase + (unsigned)(nt + 1) * (KT1 * 256u));
                bnxt.q[0] = p[0]; bnxt.q[1] = p[1];
            }
            acc1[nt] = __builtin_amdgcn_wmma_f32_16x16x32_bf16(
                false, a.v, false, bcur.v, (short)0, acc1[nt], false, false);
            bcur = bnxt;                               // renamed away by full unroll
        }
    }

    // ---- layer-1 epilogue: +bias, relu, store bf16 to LDS (lane holds column n, 8 rows) ----
    #pragma unroll
    for (int nt = 0; nt < NT1; ++nt) {
        int n = nt * 16 + mlo;
        float bb = (n < N1) ? b1[n] : 0.f;
        #pragma unroll
        for (int v = 0; v < 8; ++v) {
            if (n < N1)
                x1b[wid][v + 8 * half][n] = f2bf(fmaxf(acc1[nt][v] + bb, 0.f));
        }
    }

    // ---- layer 2: x1 @ W2^T ----
    v8f acc2[NT2];
    #pragma unroll
    for (int nt = 0; nt < NT2; ++nt)
        #pragma unroll
        for (int v = 0; v < 8; ++v) acc2[nt][v] = 0.f;

    #pragma unroll
    for (int kt = 0; kt < KT2; ++kt) {
        Frag a;
        #pragma unroll
        for (int j = 0; j < 8; ++j) {
            int ko = kt * 32 + ((j < 4) ? (half * 8 + 2 * j) : (16 + half * 8 + 2 * (j - 4)));
            a.u[j] = *(const unsigned*)&x1b[wid][mlo][ko];   // already-packed bf16 pair
        }
        const unsigned* wbase = W2p + ((unsigned)kt * 32u + lane) * 8u;
        Frag bcur, bnxt;
        bcur.q[0] = ((const u32x4*)wbase)[0];
        bcur.q[1] = ((const u32x4*)wbase)[1];
        #pragma unroll
        for (int nt = 0; nt < NT2; ++nt) {
            if (nt + 1 < NT2) {
                const u32x4* p = (const u32x4*)(wbase + (unsigned)(nt + 1) * (KT2 * 256u));
                bnxt.q[0] = p[0]; bnxt.q[1] = p[1];
            }
            acc2[nt] = __builtin_amdgcn_wmma_f32_16x16x32_bf16(
                false, a.v, false, bcur.v, (short)0, acc2[nt], false, false);
            bcur = bnxt;
        }
    }

    // ---- layer 3: relu(acc2+b2) . W3 + b3, cross-lane reduce over the 16 columns ----
    float s[8];
    #pragma unroll
    for (int v = 0; v < 8; ++v) s[v] = 0.f;
    #pragma unroll
    for (int nt = 0; nt < NT2; ++nt) {
        int n = nt * 16 + mlo;
        float bb = (n < N2) ? b2[n] : 0.f;
        float w3 = (n < N2) ? W3[n] : 0.f;
        #pragma unroll
        for (int v = 0; v < 8; ++v)
            s[v] += fmaxf(acc2[nt][v] + bb, 0.f) * w3;
    }
    float b3v = b3[0];
    #pragma unroll
    for (int v = 0; v < 8; ++v) {
        float t = s[v];
        t += __shfl_xor(t, 1, 16);
        t += __shfl_xor(t, 2, 16);
        t += __shfl_xor(t, 4, 16);
        t += __shfl_xor(t, 8, 16);
        if (mlo == 0)
            out[row0 + wid * 16 + half * 8 + v] = t + b3v;
    }
}

extern "C" void kernel_launch(void* const* d_in, const int* in_sizes, int n_in,
                              void* d_out, int out_size, void* d_ws, size_t ws_size,
                              hipStream_t stream) {
    const int*   x_cat  = (const int*)  d_in[0];
    const float* x_cont = (const float*)d_in[1];
    const float* W_cat  = (const float*)d_in[2];
    const float* b_cat  = (const float*)d_in[3];
    const float* W1     = (const float*)d_in[4];
    const float* b1     = (const float*)d_in[5];
    const float* W2     = (const float*)d_in[6];
    const float* b2     = (const float*)d_in[7];
    const float* W3     = (const float*)d_in[8];
    const float* b3     = (const float*)d_in[9];
    float* out = (float*)d_out;

    // workspace layout (~9.6 MB): Wt f32 | W1 bf16 fragments | W2 bf16 fragments
    float*    Wt  = (float*)d_ws;
    unsigned* W1p = (unsigned*)((char*)d_ws + (size_t)WT_ELEMS * sizeof(float));
    unsigned* W2p = W1p + W1P_ELEMS;

    prep_wt  <<<(WT_ELEMS  + 255) / 256, 256, 0, stream>>>(W_cat, Wt);
    prep_pack<<<(W1P_ELEMS + 255) / 256, 256, 0, stream>>>(W1, W1p, NT1, KT1, N1, K1, K1);
    prep_pack<<<(W2P_ELEMS + 255) / 256, 256, 0, stream>>>(W2, W2p, NT2, KT2, N2, K2, K2);
    fused_mlp<<<B_ROWS / TM, 128, 0, stream>>>(x_cat, x_cont, b_cat, b1, b2, W3, b3,
                                               Wt, W1p, W2p, out);
}
